// GNN_58909771432359
// MI455X (gfx1250) — compile-verified, matrix operand order checked by reading, not run on hardware
//
#include <hip/hip_runtime.h>
#include <hip/hip_bf16.h>

typedef __attribute__((ext_vector_type(16))) _Float16 v16h;
typedef __attribute__((ext_vector_type(8)))  _Float16 v8h;
typedef __attribute__((ext_vector_type(8)))  float    v8f;

#define N_NODES 50000
#define N_EDGES 320000
#define HDIM    256
#define N_GRAPH 64
#define N_EXTRA 3
#define BN_EPS  1e-5f

// ---------------- elementwise helpers ----------------

__global__ void fill_kernel(float* __restrict__ p, float v, int n) {
    int i = blockIdx.x * blockDim.x + threadIdx.x;
    if (i < n) p[i] = v;
}

__global__ void cvt_f16_kernel(const float* __restrict__ in, _Float16* __restrict__ out, int n) {
    int i = blockIdx.x * blockDim.x + threadIdx.x;
    if (i < n) out[i] = (_Float16)in[i];
}

__global__ void deg_kernel(const int* __restrict__ dst, float* __restrict__ deg, int E) {
    int e = blockIdx.x * blockDim.x + threadIdx.x;
    if (e < E) atomicAdd(&deg[dst[e]], 1.0f);
}

__global__ void rsqrt_kernel(float* __restrict__ p, int n) {
    int i = blockIdx.x * blockDim.x + threadIdx.x;
    if (i < n) p[i] = rsqrtf(p[i]);   // deg >= 1 always (self-loops)
}

// ---------------- WMMA GEMM: out[M,256] = A16[M,256] @ W16[256,256]^T ----------------
// fp16 operands loaded directly as B128 fragments (8 contiguous halves = 16B).
// One wave computes a 16x64 tile: 4 accumulators, A fragment reused 4x.
// Fragment layouts per CDNA5 ISA 7.12.2 (wave32):
//   A 16x32 f16 : lane lo=row; halves 0..7 -> K = kb + hi*8 + i ; 8..15 -> +16
//   B 32x16 f16 : lane lo=col of W^T (i.e. W row), same K packing
//   C/D 16x16 f32: vgpr r, lanes0-15 -> M = r ; lanes16-31 -> M = 8+r ; N = lo
__global__ void gemm_wmma_kernel(const _Float16* __restrict__ A16,
                                 const _Float16* __restrict__ W16,
                                 float* __restrict__ out,
                                 int nRowTiles) {
    const int K = HDIM;
    int wave = threadIdx.x >> 5;
    int lane = threadIdx.x & 31;
    int rowTile = blockIdx.x * 8 + wave;
    if (rowTile >= nRowTiles) return;          // wave-uniform: EXEC stays all-ones
    int lo = lane & 15;
    int hi = lane >> 4;
    int colBase = blockIdx.y * 64;

    const _Float16* arow = A16 + (size_t)(rowTile * 16 + lo) * K + hi * 8;
    const _Float16* wbase = W16 + (size_t)(colBase + lo) * K + hi * 8;

    v8f acc[4];
    acc[0] = (v8f){}; acc[1] = (v8f){}; acc[2] = (v8f){}; acc[3] = (v8f){};

#pragma unroll
    for (int kb = 0; kb < K; kb += 32) {
        v8h alo = *(const v8h*)(arow + kb);
        v8h ahi = *(const v8h*)(arow + kb + 16);
        v16h a = __builtin_shufflevector(alo, ahi, 0,1,2,3,4,5,6,7,8,9,10,11,12,13,14,15);
#pragma unroll
        for (int t = 0; t < 4; ++t) {
            const _Float16* wrow = wbase + (size_t)(t * 16) * K;
            v8h blo = *(const v8h*)(wrow + kb);
            v8h bhi = *(const v8h*)(wrow + kb + 16);
            v16h b = __builtin_shufflevector(blo, bhi, 0,1,2,3,4,5,6,7,8,9,10,11,12,13,14,15);
            acc[t] = __builtin_amdgcn_wmma_f32_16x16x32_f16(
                false, a, false, b, (short)0, acc[t], false, false);
        }
    }
#pragma unroll
    for (int t = 0; t < 4; ++t) {
#pragma unroll
        for (int r = 0; r < 8; ++r) {
            int row = rowTile * 16 + hi * 8 + r;
            out[(size_t)row * HDIM + colBase + t * 16 + lo] = acc[t][r];
        }
    }
}

// ---------------- edge-parallel scatter: agg[dst,:] += norm * tmp[src,:] ----------------
__global__ void scatter_kernel(const float* __restrict__ tmp,
                               const int* __restrict__ src,
                               const int* __restrict__ dst,
                               const float* __restrict__ dinv,
                               float* __restrict__ agg,
                               int E, int N) {
    int e = blockIdx.x;             // grid.x == E + N (self-loops appended)
    int c = threadIdx.x;            // 256 channels
    int s, d;
    if (e < E) { s = src[e]; d = dst[e]; }
    else       { s = d = e - E; }
    float nrm = dinv[s] * dinv[d];
    atomicAdd(&agg[(size_t)d * HDIM + c], tmp[(size_t)s * HDIM + c] * nrm);
}

// ---------------- bias -> ReLU -> BatchNorm(eval); writes fp32 + fp16 copies ----------------
__global__ void bias_relu_bn_kernel(const float* __restrict__ agg,
                                    const float* __restrict__ bias,
                                    const float* __restrict__ gamma,
                                    const float* __restrict__ beta,
                                    const float* __restrict__ rmean,
                                    const float* __restrict__ rvar,
                                    float* __restrict__ out,
                                    _Float16* __restrict__ out16, int total) {
    int i = blockIdx.x * blockDim.x + threadIdx.x;
    if (i >= total) return;
    int c = i & (HDIM - 1);
    float v = agg[i] + bias[c];
    v = fmaxf(v, 0.0f);
    v = (v - rmean[c]) * rsqrtf(rvar[c] + BN_EPS) * gamma[c] + beta[c];
    out[i] = v;
    out16[i] = (_Float16)v;
}

// ---------------- global mean pool ----------------
__global__ void pool_kernel(const float* __restrict__ h, const int* __restrict__ batch,
                            float* __restrict__ sums, float* __restrict__ cnt, int N) {
    int n = blockIdx.x;
    int c = threadIdx.x;
    int g = batch[n];
    atomicAdd(&sums[g * HDIM + c], h[(size_t)n * HDIM + c]);
    if (c == 0) atomicAdd(&cnt[g], 1.0f);
}

__global__ void pool_div_kernel(float* __restrict__ sums, const float* __restrict__ cnt, int total) {
    int i = blockIdx.x * blockDim.x + threadIdx.x;
    if (i >= total) return;
    float c = cnt[i / HDIM];
    sums[i] = sums[i] / fmaxf(c, 1.0f);
}

// ---------------- small dense linear: out[g,j] = in[g,:] . W[j,:] + b[j] ----------------
__global__ void linear_kernel(const float* __restrict__ in, const float* __restrict__ W,
                              const float* __restrict__ b, float* __restrict__ out,
                              int IN, int OUT) {
    int g = blockIdx.y;
    int j = blockIdx.x * blockDim.x + threadIdx.x;
    if (j >= OUT) return;
    const float* ir = in + g * IN;
    const float* wr = W + j * IN;
    float s = b[j];
    for (int k = 0; k < IN; ++k) s = fmaf(ir[k], wr[k], s);
    out[g * OUT + j] = s;
}

// ---------------- launch ----------------
extern "C" void kernel_launch(void* const* d_in, const int* in_sizes, int n_in,
                              void* d_out, int out_size, void* d_ws, size_t ws_size,
                              hipStream_t stream) {
    const float* x     = (const float*)d_in[0];
    const int*   eidx  = (const int*)d_in[1];       // [2,E]: row0 = src, row1 = dst
    const int*   batch = (const int*)d_in[2];
    const float* W1    = (const float*)d_in[3];
    const float* b1    = (const float*)d_in[4];
    const float* convW = (const float*)d_in[5];
    const float* convb = (const float*)d_in[6];
    const float* g1    = (const float*)d_in[7];
    const float* be1   = (const float*)d_in[8];
    const float* rm1   = (const float*)d_in[9];
    const float* rv1   = (const float*)d_in[10];
    const float* gL    = (const float*)d_in[11];
    const float* beL   = (const float*)d_in[12];
    const float* rmL   = (const float*)d_in[13];
    const float* rvL   = (const float*)d_in[14];
    const float* pW0   = (const float*)d_in[15];
    const float* pb0   = (const float*)d_in[16];
    const float* pW1   = (const float*)d_in[17];
    const float* pb1   = (const float*)d_in[18];
    const float* pW2   = (const float*)d_in[19];
    const float* pb2   = (const float*)d_in[20];

    const int* srcIdx = eidx;
    const int* dstIdx = eidx + N_EDGES;

    // workspace layout
    const size_t NH = (size_t)N_NODES * HDIM;
    float* ws   = (float*)d_ws;
    float* hbuf = ws;                               // [N, H] fp32 (aggregation target)
    float* tmp  = hbuf + NH;                        // [N, H] fp32 GEMM output
    float* dinv = tmp  + NH;                        // [N]
    float* sums = dinv + N_NODES;                   // [G, H]
    float* cnt  = sums + N_GRAPH * HDIM;            // [G]
    float* z0   = cnt  + N_GRAPH;                   // [G, 512]
    float* z1   = z0   + N_GRAPH * 512;             // [G, 256]
    _Float16* h16 = (_Float16*)(z1 + N_GRAPH * HDIM);   // [N, H] fp16 GEMM input
    _Float16* w16 = h16 + NH;                           // [4, H, H] fp16 weights

    const int nRowTiles = N_NODES / 16;             // 3125 exact
    dim3 gemmGrid((nRowTiles + 7) / 8, HDIM / 64);
    const int NHi = N_NODES * HDIM;

    // --- one-time fp16 conversions: x and all 4 weight matrices ---
    cvt_f16_kernel<<<(NHi + 255) / 256, 256, 0, stream>>>(x, h16, NHi);
    cvt_f16_kernel<<<(HDIM * HDIM + 255) / 256, 256, 0, stream>>>(W1, w16, HDIM * HDIM);
    cvt_f16_kernel<<<(N_EXTRA * HDIM * HDIM + 255) / 256, 256, 0, stream>>>(
        convW, w16 + HDIM * HDIM, N_EXTRA * HDIM * HDIM);

    // --- degrees -> dinv (self-loop contributes the initial 1.0) ---
    fill_kernel<<<(N_NODES + 255) / 256, 256, 0, stream>>>(dinv, 1.0f, N_NODES);
    deg_kernel<<<(N_EDGES + 255) / 256, 256, 0, stream>>>(dstIdx, dinv, N_EDGES);
    rsqrt_kernel<<<(N_NODES + 255) / 256, 256, 0, stream>>>(dinv, N_NODES);

    for (int layer = 0; layer < 1 + N_EXTRA; ++layer) {
        const float *bl, *ga, *be, *rm, *rv;
        if (layer == 0) { bl = b1; ga = g1; be = be1; rm = rm1; rv = rv1; }
        else {
            int i = layer - 1;
            bl = convb + i * HDIM;
            ga = gL + i * HDIM; be = beL + i * HDIM; rm = rmL + i * HDIM; rv = rvL + i * HDIM;
        }
        const _Float16* Wl16 = w16 + (size_t)layer * HDIM * HDIM;

        gemm_wmma_kernel<<<gemmGrid, 256, 0, stream>>>(h16, Wl16, tmp, nRowTiles);
        fill_kernel<<<(NHi + 255) / 256, 256, 0, stream>>>(hbuf, 0.0f, NHi);
        scatter_kernel<<<N_EDGES + N_NODES, HDIM, 0, stream>>>(tmp, srcIdx, dstIdx, dinv,
                                                               hbuf, N_EDGES, N_NODES);
        bias_relu_bn_kernel<<<(NHi + 255) / 256, 256, 0, stream>>>(
            hbuf, bl, ga, be, rm, rv, hbuf, h16, NHi);
    }

    // --- global mean pool ---
    fill_kernel<<<(N_GRAPH * HDIM + 255) / 256, 256, 0, stream>>>(sums, 0.0f, N_GRAPH * HDIM);
    fill_kernel<<<1, 256, 0, stream>>>(cnt, 0.0f, N_GRAPH);
    pool_kernel<<<N_NODES, HDIM, 0, stream>>>(hbuf, batch, sums, cnt, N_NODES);
    pool_div_kernel<<<(N_GRAPH * HDIM + 255) / 256, 256, 0, stream>>>(sums, cnt, N_GRAPH * HDIM);

    // --- post_mp: 64-row Linears (tiny; simple dot-product kernels) ---
    linear_kernel<<<dim3(4, N_GRAPH), 128, 0, stream>>>(sums, pW0, pb0, z0, HDIM, 512);
    linear_kernel<<<dim3(2, N_GRAPH), 128, 0, stream>>>(z0, pW1, pb1, z1, 512, HDIM);
    linear_kernel<<<dim3(1, N_GRAPH), 128, 0, stream>>>(z1, pW2, pb2, (float*)d_out, HDIM, 2);
}
